// EnsembleDynamicsModel_82394652606605
// MI455X (gfx1250) — matrix operand order repaired and checked by reading, not compiled, not run on hardware
//
#include <hip/hip_runtime.h>

// ---------------------------------------------------------------------------
// Ensemble dynamics MLP forward for MI455X (gfx1250, wave32, WMMA bf16).
//   x[E,B,38] -> 4x (Linear 200 + SiLU) -> Linear 62 -> mean[31], logvar[31]
// One workgroup = 1 ensemble member x 128 batch rows; whole net runs from LDS.
// ---------------------------------------------------------------------------

typedef __attribute__((ext_vector_type(16))) __bf16 v16bf;
typedef __attribute__((ext_vector_type(8)))  float  v8f;

#define ENSEMBLE  5
#define BATCH     65536
#define IN_DIM    38
#define HID       200
#define OUT2      62          // mean(31) + logvar(31)
#define OUTD      31

#define ROWS      128         // batch rows per workgroup
#define NTHREADS  256         // 8 waves (wave32)
#define ACT_K     224         // activation K stride (bf16), 7 * 32
#define W_N       208         // padded N rows of transposed weight tile (13 * 16)
#define W_K       224         // padded K cols of transposed weight tile

#define MEAN_TOTAL ((size_t)ENSEMBLE * BATCH * OUTD)

// Fast SiLU: x * rcp(1+exp(-x)) -> v_exp_f32 + v_rcp_f32 + mul (no IEEE div expansion)
__device__ __forceinline__ float silu_f(float x) {
    return x * __builtin_amdgcn_rcpf(1.0f + __expf(-x));
}
__device__ __forceinline__ float softplus_f(float x) {
    return (x > 15.0f) ? x : __logf(1.0f + __expf(x));
}

struct Frag32B { uint4 a, b; };   // 32 bytes == v16bf

// A-fragment (16x32 bf16, contiguous K, matching B):
//   lanes 0-15  : M=lane,    K = kb + 0..15
//   lanes 16-31 : M=lane-16, K = kb + 16..31
__device__ __forceinline__ v16bf load_afrag(const __bf16* act, int row, int kb, int half) {
    const uint4* p = (const uint4*)(act + row * ACT_K + kb + half * 16);
    Frag32B u; u.a = p[0]; u.b = p[1];          // +0..7 and +8..15 bf16
    return __builtin_bit_cast(v16bf, u);
}

// B-fragment (32x16 bf16) from transposed weights wT[N][W_K]:
//   lanes 0-15  : N=lane,    K = kb + 0..15 (contiguous)
//   lanes 16-31 : N=lane-16, K = kb + 16..31
__device__ __forceinline__ v16bf load_bfrag(const __bf16* wT, int n, int kb) {
    const uint4* p = (const uint4*)(wT + n * W_K + kb);
    Frag32B u; u.a = p[0]; u.b = p[1];
    return __builtin_bit_cast(v16bf, u);
}

// Prefetch a global range into cache (global_prefetch_b8), one 128B line/thread step.
__device__ __forceinline__ void prefetch_range(const float* p, size_t bytes, int tid) {
    const char* c = (const char*)p;
    for (size_t off = (size_t)tid * 128; off < bytes; off += (size_t)NTHREADS * 128)
        __builtin_prefetch(c + off, 0, 1);
}

enum ZeroMode { ZERO_ALL, ZERO_NONE, ZERO_TAIL };   // TAIL: rows [OUT2, W_N)

// Stage one layer's weights transposed into LDS as bf16, bias as f32.
__device__ __forceinline__ void stage_weights(__bf16* s_wT, float* s_bias,
                                              const float* __restrict__ w,
                                              const float* __restrict__ b,
                                              int K, int N, int e, int tid,
                                              ZeroMode zm) {
    __syncthreads();                                   // previous layer done with s_wT
    if (zm == ZERO_ALL) {
        for (int i = tid; i < (W_N * W_K) / 2; i += NTHREADS)
            ((unsigned int*)s_wT)[i] = 0u;
        __syncthreads();                               // zero overlaps the fill region
    } else if (zm == ZERO_TAIL) {                      // rows [OUT2,W_N): disjoint from fill
        unsigned int* z = (unsigned int*)(s_wT + OUT2 * W_K);
        for (int i = tid; i < ((W_N - OUT2) * W_K) / 2; i += NTHREADS)
            z[i] = 0u;
    }
    const size_t woff = (size_t)e * K * N;
    for (int i = tid; i < K * N; i += NTHREADS) {
        int k = i / N, n = i - k * N;
        s_wT[n * W_K + k] = (__bf16)w[woff + i];       // transpose: [K][N] -> [N][K]
    }
    for (int n = tid; n < W_N; n += NTHREADS)
        s_bias[n] = (n < N) ? b[(size_t)e * N + n] : 0.0f;
    __syncthreads();
}

// Hidden layer: act(bf16, in LDS) x wT -> SiLU -> act (in place; rows are wave-private).
template <int KT, int NT>
__device__ __forceinline__ void layer_hidden(__bf16* act, const __bf16* s_wT,
                                             const float* s_bias, int wave, int lane) {
    const int half = lane >> 4, l16 = lane & 15;
    const int rbase = wave * 16;

    v16bf a[KT];
#pragma unroll
    for (int kk = 0; kk < KT; ++kk)
        a[kk] = load_afrag(act, rbase + l16, kk * 32, half);

    for (int nt = 0; nt < NT; ++nt) {
        v8f acc = {};
#pragma unroll
        for (int kk = 0; kk < KT; ++kk) {
            v16bf bf = load_bfrag(s_wT, nt * 16 + l16, kk * 32 + half * 16);
            acc = __builtin_amdgcn_wmma_f32_16x16x32_bf16(
                false, a[kk], false, bf, (short)0, acc, false, false);
        }
        const int   n  = nt * 16 + l16;
        const float bv = s_bias[n];
        // C/D layout: VGPR j holds (M = j + half*8, N = n)
#pragma unroll
        for (int j = 0; j < 8; ++j) {
            float v = silu_f(acc[j] + bv);
            act[(rbase + half * 8 + j) * ACT_K + n] = (__bf16)v;
        }
    }
}

// Output layer: GEMM + bias, then mean / softplus-clamped logvar straight to HBM.
template <int KT>
__device__ __forceinline__ void layer_out(const __bf16* act, const __bf16* s_wT,
                                          const float* s_bias, float* __restrict__ out,
                                          const float* __restrict__ maxlv,
                                          const float* __restrict__ minlv,
                                          int e, int row0, int wave, int lane) {
    const int half = lane >> 4, l16 = lane & 15;
    const int rbase = wave * 16;

    v16bf a[KT];
#pragma unroll
    for (int kk = 0; kk < KT; ++kk)
        a[kk] = load_afrag(act, rbase + l16, kk * 32, half);

    for (int nt = 0; nt < 4; ++nt) {          // 4 * 16 = 64 >= 62 output cols
        v8f acc = {};
#pragma unroll
        for (int kk = 0; kk < KT; ++kk) {
            v16bf bf = load_bfrag(s_wT, nt * 16 + l16, kk * 32 + half * 16);
            acc = __builtin_amdgcn_wmma_f32_16x16x32_bf16(
                false, a[kk], false, bf, (short)0, acc, false, false);
        }
        const int n = nt * 16 + l16;
        if (n >= OUT2) continue;
        const float bv = s_bias[n];
#pragma unroll
        for (int j = 0; j < 8; ++j) {
            const int row = row0 + rbase + half * 8 + j;
            float v = acc[j] + bv;
            if (n < OUTD) {
                out[((size_t)e * BATCH + row) * OUTD + n] = v;
            } else {
                const int   c  = n - OUTD;
                const float mx = maxlv[c];
                const float mn = minlv[c];
                float lv = mx - softplus_f(mx - v);
                lv = mn + softplus_f(lv - mn);
                out[MEAN_TOTAL + ((size_t)e * BATCH + row) * OUTD + c] = lv;
            }
        }
    }
}

__global__ __launch_bounds__(NTHREADS)
void ensemble_mlp_kernel(const float* __restrict__ x,
                         const float* __restrict__ w0, const float* __restrict__ b0,
                         const float* __restrict__ w1, const float* __restrict__ b1,
                         const float* __restrict__ w2, const float* __restrict__ b2,
                         const float* __restrict__ w3, const float* __restrict__ b3,
                         const float* __restrict__ w4, const float* __restrict__ b4,
                         const float* __restrict__ maxlv,
                         const float* __restrict__ minlv,
                         float* __restrict__ out) {
    __shared__ __bf16 s_act[ROWS * ACT_K];   //  57,344 B
    __shared__ __bf16 s_wT[W_N * W_K];       //  93,184 B
    __shared__ float  s_bias[W_N];           //     832 B   (~151 KB of 320 KB/WGP)

    const int tiles = BATCH / ROWS;
    const int e     = blockIdx.x / tiles;
    const int tile  = blockIdx.x - e * tiles;
    const int row0  = tile * ROWS;
    const int tid   = threadIdx.x;
    const int wave  = tid >> 5;
    const int lane  = tid & 31;

    const size_t eHH = (size_t)e * HID * HID;

    // Stage input slab: f32 -> bf16, zero-pad K in [38, 224). Pad cols beyond
    // 208 are never rewritten; cols [200,208) get silu(0+0)=0 each layer.
    const size_t xbase = ((size_t)e * BATCH + row0) * IN_DIM;
    for (int i = tid; i < ROWS * ACT_K; i += NTHREADS) {
        int r = i / ACT_K, c = i - r * ACT_K;
        float v = (c < IN_DIM) ? x[xbase + (size_t)r * IN_DIM + c] : 0.0f;
        s_act[i] = (__bf16)v;
    }
    // (stage_weights' barriers also order s_act for layer 0 fragment loads)

    stage_weights(s_wT, s_bias, w0, b0, IN_DIM, HID, e, tid, ZERO_ALL);
    prefetch_range(w1 + eHH, (size_t)HID * HID * 4, tid);       // hide next layer fetch
    layer_hidden<2, 13>(s_act, s_wT, s_bias, wave, lane);       // K 38->64,  N 200->208

    stage_weights(s_wT, s_bias, w1, b1, HID, HID, e, tid, ZERO_NONE);
    prefetch_range(w2 + eHH, (size_t)HID * HID * 4, tid);
    layer_hidden<7, 13>(s_act, s_wT, s_bias, wave, lane);

    stage_weights(s_wT, s_bias, w2, b2, HID, HID, e, tid, ZERO_NONE);
    prefetch_range(w3 + eHH, (size_t)HID * HID * 4, tid);
    layer_hidden<7, 13>(s_act, s_wT, s_bias, wave, lane);

    stage_weights(s_wT, s_bias, w3, b3, HID, HID, e, tid, ZERO_NONE);
    prefetch_range(w4 + (size_t)e * HID * OUT2, (size_t)HID * OUT2 * 4, tid);
    layer_hidden<7, 13>(s_act, s_wT, s_bias, wave, lane);

    stage_weights(s_wT, s_bias, w4, b4, HID, OUT2, e, tid, ZERO_TAIL);
    layer_out<7>(s_act, s_wT, s_bias, out, maxlv, minlv, e, row0, wave, lane);
}

extern "C" void kernel_launch(void* const* d_in, const int* in_sizes, int n_in,
                              void* d_out, int out_size, void* d_ws, size_t ws_size,
                              hipStream_t stream) {
    // setup_inputs() order: w0,b0,w1,b1,w2,b2,w3,b3,w4,b4,x,max_logvar,min_logvar
    const float* w0 = (const float*)d_in[0];
    const float* b0 = (const float*)d_in[1];
    const float* w1 = (const float*)d_in[2];
    const float* b1 = (const float*)d_in[3];
    const float* w2 = (const float*)d_in[4];
    const float* b2 = (const float*)d_in[5];
    const float* w3 = (const float*)d_in[6];
    const float* b3 = (const float*)d_in[7];
    const float* w4 = (const float*)d_in[8];
    const float* b4 = (const float*)d_in[9];
    const float* x  = (const float*)d_in[10];
    const float* mx = (const float*)d_in[11];
    const float* mn = (const float*)d_in[12];
    float* out = (float*)d_out;

    dim3 grid(ENSEMBLE * (BATCH / ROWS));   // 5 * 512 = 2560 workgroups
    dim3 block(NTHREADS);
    ensemble_mlp_kernel<<<grid, block, 0, stream>>>(
        x, w0, b0, w1, b1, w2, b2, w3, b3, w4, b4, mx, mn, out);
}
